// Sender_LSTM_18657337934621
// MI455X (gfx1250) — matrix operand order repaired and checked by reading, not compiled, not run on hardware
//
#include <hip/hip_runtime.h>
#include <hip/hip_bf16.h>
#include <stdint.h>
#include <math.h>

// ---------------------------------------------------------------------------
// Types for CDNA5 WMMA / TDM
// ---------------------------------------------------------------------------
typedef __bf16 bf16;
typedef __attribute__((ext_vector_type(16))) __bf16 v16bf;
typedef __attribute__((ext_vector_type(8)))  float  v8f;
typedef __attribute__((ext_vector_type(4))) unsigned int u32x4;
typedef __attribute__((ext_vector_type(4))) int i32x4;
typedef __attribute__((ext_vector_type(8))) int i32x8;

#define DEV static __device__ __forceinline__

// Tensor Data Mover availability. This toolchain (clang-23/therock-10.0)
// exposes the 6-arg form: (u32x4 g0, i32x8 g1, i32x4, i32x4, i32x8, i32 cpol).
#if defined(__has_builtin)
#  if __has_builtin(__builtin_amdgcn_tensor_load_to_lds) && \
      __has_builtin(__builtin_amdgcn_s_wait_tensorcnt)
#    define HAVE_TDM 1
#  endif
#endif
#ifndef HAVE_TDM
#  define HAVE_TDM 0
#endif

DEV v8f wmma_bf16(v16bf a, v16bf b, v8f c) {
  // D(16x16,f32) = A(16x32,bf16) x B(32x16,bf16) + C
  return __builtin_amdgcn_wmma_f32_16x16x32_bf16(
      /*neg_a=*/false, a, /*neg_b=*/false, b,
      /*c_mod=*/(short)0, c, /*reuse_a=*/false, /*reuse_b=*/false);
}

// A-fragment (16x32 bf16, row-major source with leading dim lda):
// lane: row = row0 + (lane&15), half = lane>>4
//  VGPR 0..3  hold K = half*8 + 0..7       (contiguous 8 elems)
//  VGPR 4..7  hold K = 16 + half*8 + 0..7  (contiguous 8 elems)
DEV v16bf load_a_frag(const bf16* __restrict__ A, int lda, int row0, int kb, int lane) {
  const int row  = row0 + (lane & 15);
  const int half = lane >> 4;
  const bf16* p = A + (size_t)row * lda + kb + half * 8;
  v16bf a;
#pragma unroll
  for (int j = 0; j < 8; ++j) a[j] = p[j];
#pragma unroll
  for (int j = 0; j < 8; ++j) a[8 + j] = p[16 + j];
  return a;
}

// B-fragment (32x16 bf16) from an [N][K] tile (col n contiguous in K):
// lane: n = lane&15 fixed, elems 0..7 hold K=half*8+0..7, elems 8..15 hold K=16+half*8+0..7
DEV v16bf load_b_frag(const bf16* tile_row /* tile + n*ldk */, int kb, int half) {
  const bf16* p = tile_row + kb + half * 8;
  v16bf b;
#pragma unroll
  for (int j = 0; j < 8; ++j) b[j] = p[j];
#pragma unroll
  for (int j = 0; j < 8; ++j) b[8 + j] = p[16 + j];
  return b;
}

DEV float sigmoidf_(float x) { return 1.0f / (1.0f + __expf(-x)); }

// Branch-free tanh: clamp then (e^{2x}-1)/(e^{2x}+1); single v_exp_f32.
DEV float tanhf_(float x) {
  x = fminf(fmaxf(x, -15.0f), 15.0f);
  float e = __expf(2.0f * x);
  return (e - 1.0f) / (e + 1.0f);
}

DEV float gumbelf(uint32_t msg, uint32_t row, uint32_t col) {
  uint32_t x = 0x243F6A88u ^ (msg * 0x9E3779B9u) ^ (row * 0x85EBCA6Bu) ^ (col * 0xC2B2AE35u);
  x ^= x >> 16; x *= 0x7FEB352Du;
  x ^= x >> 15; x *= 0x846CA68Bu;
  x ^= x >> 16;
  float u = (float)(x >> 8) * (1.0f / 16777216.0f) + 1.0e-12f;
  return -__logf(-__logf(u));
}

#if HAVE_TDM
// TDM: DMA a 16-row x 512-elem bf16 tile ([N][K], K contiguous, stride 512)
// from global into LDS. Issue from ONE wave only (wave-level instruction).
// D# per CDNA5 ISA; tracked with TENSORcnt.
DEV void tdm_load_tile16x512(const bf16* gsrc, void* lds_dst) {
  uint64_t ga  = (uint64_t)(uintptr_t)gsrc;
  uint32_t lds = (uint32_t)(uintptr_t)lds_dst;  // low 32 bits = LDS byte offset
  u32x4 g0;
  g0[0] = 1u;                                    // count=1 (valid user D#)
  g0[1] = lds;                                   // lds_addr
  g0[2] = (uint32_t)ga;                          // global_addr[31:0]
  g0[3] = (uint32_t)((ga >> 32) & 0x01FFFFFFu)   // global_addr[56:32]
          | (2u << 30);                          // type = 2 ("image")
  i32x8 g1;
  g1[0] = (int)(1u << 16);     // workgroup_mask=0 | data_size=1 (2 bytes)
  g1[1] = (int)(512u << 16);   // tensor_dim0[15:0] = 512
  g1[2] = (int)(16u << 16);    // tensor_dim0 hi=0 | tensor_dim1[15:0] = 16
  g1[3] = (int)(512u << 16);   // tensor_dim1 hi=0 | tile_dim0 = 512
  g1[4] = 16;                  // tile_dim1 = 16 | tile_dim2 = 0
  g1[5] = 512;                 // tensor_dim0_stride[31:0] = 512
  g1[6] = 0;                   // stride hi | tensor_dim1_stride lo
  g1[7] = 0;
  i32x4 z4 = {0, 0, 0, 0};     // groups 2/3 unused (<=2D tensor)
  i32x8 z8 = {0, 0, 0, 0, 0, 0, 0, 0};
  __builtin_amdgcn_tensor_load_to_lds(g0, g1, z4, z4, z8, 0);
}
#endif

// ---------------------------------------------------------------------------
// Utility kernels
// ---------------------------------------------------------------------------
__global__ void k_cast_bf16(const float* __restrict__ s, bf16* __restrict__ d, int n) {
  int i = blockIdx.x * blockDim.x + threadIdx.x;
  if (i < n) d[i] = (bf16)s[i];
}

// W: K x N (f32, row-major) ->  WT: N x K (bf16)
__global__ void k_cast_transpose(const float* __restrict__ W, bf16* __restrict__ WT,
                                 int K, int N) {
  int i = blockIdx.x * blockDim.x + threadIdx.x;
  if (i >= K * N) return;
  int k = i / N, n = i - k * N;
  WT[(size_t)n * K + k] = (bf16)W[i];
}

__global__ void k_zero32(uint32_t* __restrict__ p, int n) {
  int i = blockIdx.x * blockDim.x + threadIdx.x;
  if (i < n) p[i] = 0u;
}

__global__ void k_copyf(const float* __restrict__ s, float* __restrict__ d, int n) {
  int i = blockIdx.x * blockDim.x + threadIdx.x;
  if (i < n) d[i] = s[i];
}

// ---------------------------------------------------------------------------
// Fused LSTM cell:  z = X@Wx + Hin@Wh + b ; gates ; update c, h
//   X   : B x K1 bf16, leading dim ldx
//   WxT : (4H) x K1 bf16   (transposed, col-contiguous-in-K)
//   Hin : B x H bf16
//   WhT : (4H) x H bf16
// Grid: (B/128, H/16), 256 threads (8 waves; wave w -> rows base + w*16).
// ---------------------------------------------------------------------------
__global__ void __launch_bounds__(256)
k_lstm_cell(const bf16* __restrict__ X, int ldx, int K1,
            const bf16* __restrict__ WxT,
            const bf16* __restrict__ Hin,
            const bf16* __restrict__ WhT,
            const float* __restrict__ bias,
            float* __restrict__ C,
            float* __restrict__ Hf32,   // may be null
            bf16*  __restrict__ Hbf,
            int H) {
  __shared__ __align__(16) bf16 tb[4 * 16 * 64]; // [gate*16+n][64] K-chunk tile

  const int tid  = threadIdx.x;
  const int wave = tid >> 5;
  const int lane = tid & 31;
  const int row0 = blockIdx.x * 128 + wave * 16;
  const int col0 = blockIdx.y * 16;     // column slice within H
  const int half = lane >> 4;
  const int nloc = lane & 15;

  v8f acc[4];
#pragma unroll
  for (int g = 0; g < 4; ++g) { v8f z = {}; acc[g] = z; }

#pragma unroll 1
  for (int phase = 0; phase < 2; ++phase) {
    const bf16* A   = phase ? Hin : X;
    const int   lda = phase ? H : ldx;
    const int   K   = phase ? H : K1;
    const bf16* WT  = phase ? WhT : WxT;

#pragma unroll 1
    for (int kb = 0; kb < K; kb += 64) {
      __syncthreads();
      {
        // cooperative load: 64 rows (gate*16+n) x 64 bf16 (128B each)
        int r    = tid >> 2;   // 0..63
        int part = tid & 3;    // 0..3 -> 32B each
        int g = r >> 4, n = r & 15;
        const uint4* src = (const uint4*)(WT + (size_t)(g * H + col0 + n) * K + kb);
        uint4* dst = (uint4*)(tb + r * 64);
        dst[part * 2 + 0] = src[part * 2 + 0];
        dst[part * 2 + 1] = src[part * 2 + 1];
      }
      __syncthreads();
#pragma unroll
      for (int sub = 0; sub < 2; ++sub) {
        v16bf a = load_a_frag(A, lda, row0, kb + sub * 32, lane);
#pragma unroll
        for (int g = 0; g < 4; ++g) {
          v16bf b = load_b_frag(tb + (g * 16 + nloc) * 64, sub * 32, half);
          acc[g] = wmma_bf16(a, b, acc[g]);
        }
      }
    }
  }

  // Gate math. C/D layout: elem r of lane -> row = r + 8*half, col = lane&15.
  const int colg = col0 + nloc;
  const float bi  = bias[0 * H + colg];
  const float bff = bias[1 * H + colg];
  const float bg  = bias[2 * H + colg];
  const float bo  = bias[3 * H + colg];

#pragma unroll
  for (int r = 0; r < 8; ++r) {
    const int row = row0 + r + 8 * half;
    const size_t idx = (size_t)row * H + colg;
    float iv = sigmoidf_(acc[0][r] + bi);
    float fv = sigmoidf_(acc[1][r] + bff);
    float gv = tanhf_(acc[2][r] + bg);
    float ov = sigmoidf_(acc[3][r] + bo);
    float cn = fv * C[idx] + iv * gv;
    float hn = ov * tanhf_(cn);
    C[idx] = cn;
    if (Hf32) Hf32[idx] = hn;
    Hbf[idx] = (bf16)hn;
  }
}

// ---------------------------------------------------------------------------
// Decoder tile pipeline (shared by both passes):
//   double-buffered 16-col x 512-K bf16 W tile in LDS, staged by TDM DMA
//   (wave 0 issues tensor_load_to_lds for tile i+1 while all waves compute
//   tile i); fallback = synchronous cooperative b128 staging.
// ---------------------------------------------------------------------------
#if HAVE_TDM
#define DEC_STAGE_PROLOGUE(WdT, tb, wave)                                      \
  if ((wave) == 0) tdm_load_tile16x512((WdT), &(tb)[0][0]);
#define DEC_STAGE(WdT, tb, i, ntiles, cur, tid, wave)                          \
  __builtin_amdgcn_s_wait_tensorcnt(0);                                        \
  __syncthreads();                                                             \
  if ((wave) == 0 && ((i) + 1) < (ntiles))                                     \
    tdm_load_tile16x512((WdT) + ((size_t)((i) + 1) * 16) * 512,                \
                        &(tb)[((i) + 1) & 1][0]);
#else
#define DEC_STAGE_PROLOGUE(WdT, tb, wave)
#define DEC_STAGE(WdT, tb, i, ntiles, cur, tid, wave)                          \
  __syncthreads();                                                             \
  {                                                                            \
    int r_ = (tid) >> 4, part_ = (tid) & 15;                                   \
    const uint4* src_ = (const uint4*)((WdT) + (size_t)((i) * 16 + r_) * 512); \
    uint4* dst_ = (uint4*)((cur) + r_ * 512);                                  \
    _Pragma("unroll")                                                          \
    for (int q_ = 0; q_ < 4; ++q_) dst_[part_ * 4 + q_] = src_[part_ * 4 + q_];\
  }                                                                            \
  __syncthreads();
#endif

// ---------------------------------------------------------------------------
// Decoder pass 1: logits streamed, online softmax stats -> logZ, entropy.
// Grid: B/128 blocks, 256 threads (8 waves x 16 rows). K = 512 fixed.
// ---------------------------------------------------------------------------
__global__ void __launch_bounds__(256)
k_dec_stats(const bf16* __restrict__ Hb,      // B x 512 bf16
            const bf16* __restrict__ WdT,     // V x 512 bf16
            const float* __restrict__ bd,     // V
            float* __restrict__ logZ,         // B
            float* __restrict__ ent_out,      // TM x B
            int B, int V, int TM) {
  __shared__ __align__(16) bf16 tb[2][16 * 512]; // double-buffered vocab tile
  __shared__ float red_m[256][8];
  __shared__ float red_s[256][8];
  __shared__ float red_u[256][8];

  const int tid  = threadIdx.x;
  const int wave = tid >> 5;
  const int lane = tid & 31;
  const int half = lane >> 4;
  const int nloc = lane & 15;
  const int row0 = blockIdx.x * 128 + wave * 16;
  const int ntiles = V / 16;

  // Preload A fragments for this wave's 16 rows (full K=512).
  v16bf af[16];
#pragma unroll
  for (int kc = 0; kc < 16; ++kc) af[kc] = load_a_frag(Hb, 512, row0, kc * 32, lane);

  float m[8], s[8], u[8];
#pragma unroll
  for (int r = 0; r < 8; ++r) { m[r] = -INFINITY; s[r] = 0.0f; u[r] = 0.0f; }

  DEC_STAGE_PROLOGUE(WdT, tb, wave)

#pragma unroll 1
  for (int i = 0; i < ntiles; ++i) {
    bf16* cur = &tb[i & 1][0];
    DEC_STAGE(WdT, tb, i, ntiles, cur, tid, wave)

    v8f acc = {};
#pragma unroll
    for (int kc = 0; kc < 16; ++kc) {
      v16bf b = load_b_frag(cur + nloc * 512, kc * 32, half);
      acc = wmma_bf16(af[kc], b, acc);
    }
    const float bcol = bd[i * 16 + nloc];
#pragma unroll
    for (int r = 0; r < 8; ++r) {
      float l  = acc[r] + bcol;
      float mn = fmaxf(m[r], l);
      float sc = __expf(m[r] - mn);
      float el = __expf(l - mn);
      s[r] = s[r] * sc + el;
      u[r] = u[r] * sc + l * el;
      m[r] = mn;
    }
  }

#pragma unroll
  for (int r = 0; r < 8; ++r) { red_m[tid][r] = m[r]; red_s[tid][r] = s[r]; red_u[tid][r] = u[r]; }
  __syncthreads();

  if (tid < 128) {
    int w = tid >> 4, t = tid & 15;
    int h2 = t >> 3, r = t & 7;
    float M = -INFINITY, S = 0.0f, U = 0.0f;
    for (int ln = 0; ln < 16; ++ln) {
      int st = w * 32 + h2 * 16 + ln;
      float m2 = red_m[st][r], s2 = red_s[st][r], u2 = red_u[st][r];
      float Mn  = fmaxf(M, m2);
      float scA = __expf(M - Mn);
      float scB = __expf(m2 - Mn);
      S = S * scA + s2 * scB;
      U = U * scA + u2 * scB;
      M = Mn;
    }
    float lz = M + __logf(S);
    float en = lz - U / S;
    int grow = blockIdx.x * 128 + w * 16 + t;
    logZ[grow] = lz;
    for (int mm = 0; mm < TM; ++mm) ent_out[mm * B + grow] = en;
  }
}

// ---------------------------------------------------------------------------
// Decoder pass 2: Gumbel-max categorical sampling for one message per block-y.
// Grid: (B/128, TM), 256 threads.
// ---------------------------------------------------------------------------
__global__ void __launch_bounds__(256)
k_dec_sample(const bf16* __restrict__ Hb,
             const bf16* __restrict__ WdT,
             const float* __restrict__ bd,
             const float* __restrict__ logZ,
             int*   __restrict__ msg_out,   // B x (2*TM)
             float* __restrict__ lp_out,    // TM x B
             int B, int V, int TM) {
  __shared__ __align__(16) bf16 tb[2][16 * 512];
  __shared__ float red_v[256][8];
  __shared__ float red_l[256][8];
  __shared__ int   red_i[256][8];

  const int tid  = threadIdx.x;
  const int wave = tid >> 5;
  const int lane = tid & 31;
  const int half = lane >> 4;
  const int nloc = lane & 15;
  const int row0 = blockIdx.x * 128 + wave * 16;
  const int msg  = blockIdx.y;
  const int ntiles = V / 16;

  v16bf af[16];
#pragma unroll
  for (int kc = 0; kc < 16; ++kc) af[kc] = load_a_frag(Hb, 512, row0, kc * 32, lane);

  float bv[8], bl[8];
  int   bi_[8];
#pragma unroll
  for (int r = 0; r < 8; ++r) { bv[r] = -INFINITY; bl[r] = 0.0f; bi_[r] = 0; }

  DEC_STAGE_PROLOGUE(WdT, tb, wave)

#pragma unroll 1
  for (int i = 0; i < ntiles; ++i) {
    bf16* cur = &tb[i & 1][0];
    DEC_STAGE(WdT, tb, i, ntiles, cur, tid, wave)

    v8f acc = {};
#pragma unroll
    for (int kc = 0; kc < 16; ++kc) {
      v16bf b = load_b_frag(cur + nloc * 512, kc * 32, half);
      acc = wmma_bf16(af[kc], b, acc);
    }
    const float bcol = bd[i * 16 + nloc];
    const int   col  = i * 16 + nloc;
#pragma unroll
    for (int r = 0; r < 8; ++r) {
      const int row = row0 + r + 8 * half;
      float l = acc[r] + bcol;
      float v = l + gumbelf((uint32_t)msg, (uint32_t)row, (uint32_t)col);
      if (v > bv[r]) { bv[r] = v; bl[r] = l; bi_[r] = col; }
    }
  }

#pragma unroll
  for (int r = 0; r < 8; ++r) { red_v[tid][r] = bv[r]; red_l[tid][r] = bl[r]; red_i[tid][r] = bi_[r]; }
  __syncthreads();

  if (tid < 128) {
    int w = tid >> 4, t = tid & 15;
    int h2 = t >> 3, r = t & 7;
    float V_ = -INFINITY, L_ = 0.0f;
    int   I_ = 0;
    for (int ln = 0; ln < 16; ++ln) {
      int st = w * 32 + h2 * 16 + ln;
      float v2 = red_v[st][r];
      if (v2 > V_) { V_ = v2; L_ = red_l[st][r]; I_ = red_i[st][r]; }
    }
    int grow = blockIdx.x * 128 + w * 16 + t;
    msg_out[(size_t)grow * (2 * TM) + msg] = I_;
    lp_out[msg * B + grow] = L_ - logZ[grow];
  }
}

// ---------------------------------------------------------------------------
// Host launcher
// ---------------------------------------------------------------------------
extern "C" void kernel_launch(void* const* d_in, const int* in_sizes, int n_in,
                              void* d_out, int out_size, void* d_ws, size_t ws_size,
                              hipStream_t stream) {
  (void)in_sizes; (void)n_in; (void)out_size; (void)ws_size;

  constexpr int B = 4096, T = 4, E = 256, H = 512, V = 32000, TM = 15;
  constexpr int H4 = 4 * H;

  const float* inp = (const float*)d_in[0];
  const float* Wx0 = (const float*)d_in[1];
  const float* Wh0 = (const float*)d_in[2];
  const float* b0  = (const float*)d_in[3];
  const float* Wx1 = (const float*)d_in[4];
  const float* Wh1 = (const float*)d_in[5];
  const float* b1  = (const float*)d_in[6];
  const float* Wd  = (const float*)d_in[7];
  const float* bd  = (const float*)d_in[8];

  // Workspace carve-up (256B aligned)
  char* ws = (char*)d_ws;
  size_t off = 0;
  auto carve = [&](size_t bytes) { char* p = ws + off; off += (bytes + 255) & ~(size_t)255; return p; };

  bf16* tWx0 = (bf16*)carve((size_t)H4 * E * 2);
  bf16* tWh0 = (bf16*)carve((size_t)H4 * H * 2);
  bf16* tWx1 = (bf16*)carve((size_t)H4 * H * 2);
  bf16* tWh1 = (bf16*)carve((size_t)H4 * H * 2);
  bf16* tWd  = (bf16*)carve((size_t)V * H * 2);
  bf16* binp = (bf16*)carve((size_t)B * T * E * 2);
  float* c0  = (float*)carve((size_t)B * H * 4);
  float* c1  = (float*)carve((size_t)B * H * 4);
  float* h0f = (float*)carve((size_t)B * H * 4);
  bf16* bh0a = (bf16*)carve((size_t)B * H * 2);
  bf16* bh0b = (bf16*)carve((size_t)B * H * 2);
  bf16* bh1a = (bf16*)carve((size_t)B * H * 2);
  bf16* bh1b = (bf16*)carve((size_t)B * H * 2);
  float* lZ  = (float*)carve((size_t)B * 4);

  float* out = (float*)d_out;
  // Output layout (flat, reference return order):
  const size_t off_lp  = (size_t)B * 2 * TM;      // (TM, B) f32
  const size_t off_ent = off_lp + (size_t)TM * B; // (TM, B) f32
  const size_t off_st  = off_ent + (size_t)TM * B;// (2, B, H) f32

  auto nb = [](long n) { return (int)((n + 255) / 256); };

  // Weight precision conversion (f32 -> bf16, [N][K] transposed layouts)
  k_cast_transpose<<<nb((long)E * H4), 256, 0, stream>>>(Wx0, tWx0, E, H4);
  k_cast_transpose<<<nb((long)H * H4), 256, 0, stream>>>(Wh0, tWh0, H, H4);
  k_cast_transpose<<<nb((long)H * H4), 256, 0, stream>>>(Wx1, tWx1, H, H4);
  k_cast_transpose<<<nb((long)H * H4), 256, 0, stream>>>(Wh1, tWh1, H, H4);
  k_cast_transpose<<<nb((long)H * V),  256, 0, stream>>>(Wd,  tWd,  H, V);
  k_cast_bf16<<<nb((long)B * T * E), 256, 0, stream>>>(inp, binp, B * T * E);

  // Zero initial states (c0, c1, h-in buffers) and the message output region
  k_zero32<<<nb((long)B * H), 256, 0, stream>>>((uint32_t*)c0, B * H);
  k_zero32<<<nb((long)B * H), 256, 0, stream>>>((uint32_t*)c1, B * H);
  k_zero32<<<nb((long)B * H / 2), 256, 0, stream>>>((uint32_t*)bh0a, B * H / 2);
  k_zero32<<<nb((long)B * H / 2), 256, 0, stream>>>((uint32_t*)bh1a, B * H / 2);
  k_zero32<<<nb((long)B * 2 * TM), 256, 0, stream>>>((uint32_t*)d_out, B * 2 * TM);

  // 2-layer LSTM scan over T timesteps (h double-buffered)
  const bf16* h0r = bh0a; bf16* h0w = bh0b;
  const bf16* h1r = bh1a; bf16* h1w = bh1b;
  const dim3 lgrid(B / 128, H / 16);
  for (int t = 0; t < T; ++t) {
    k_lstm_cell<<<lgrid, 256, 0, stream>>>(binp + (size_t)t * E, T * E, E, tWx0,
                                           h0r, tWh0, b0, c0, h0f, h0w, H);
    k_lstm_cell<<<lgrid, 256, 0, stream>>>(h0w, H, H, tWx1,
                                           h1r, tWh1, b1, c1, nullptr, h1w, H);
    const bf16* tmp0 = h0r; h0r = h0w; h0w = (bf16*)tmp0;
    const bf16* tmp1 = h1r; h1r = h1w; h1w = (bf16*)tmp1;
  }

  // Decoder: fused GEMM + online log-softmax stats, then Gumbel-max sampling
  k_dec_stats<<<B / 128, 256, 0, stream>>>(h1r, tWd, bd, lZ, out + off_ent, B, V, TM);
  k_dec_sample<<<dim3(B / 128, TM), 256, 0, stream>>>(h1r, tWd, bd, lZ,
                                                      (int*)d_out, out + off_lp, B, V, TM);

  // state_h = stack([h0, c0])
  k_copyf<<<nb((long)B * H), 256, 0, stream>>>(h0f, out + off_st, B * H);
  k_copyf<<<nb((long)B * H), 256, 0, stream>>>(c0, out + off_st + (size_t)B * H, B * H);
}
